// EuclideanCodebook_87531433492804
// MI455X (gfx1250) — compile-verified, matrix operand order checked by reading, not run on hardware
//
#include <hip/hip_runtime.h>

// ---------------------------------------------------------------------------
// VQ-VAE Euclidean codebook update for gfx1250 (MI455X, wave32, WMMA).
//
// Outputs (flat, in order): quantize[48000*256], codes[48000],
// new_cluster_size[2048], new_embed_avg[2048*256], new_embed[2048*256].
//
// v2.1: B tiles staged once per block into LDS via async global->LDS copies
// (double buffered, ASYNCcnt-tracked), consumed by all 8 waves via ds loads.
// ---------------------------------------------------------------------------

#define DIMD 256
#define KCODES 2048
#define NROWS 48000
#define DECAY 0.99f
#define OMD 0.01f            // 1 - DECAY
#define EPSV 1e-5f
#define THRESH 2.0f
#define BPITCH 264           // LDS row pitch in bf16 units (528 B): bank-conflict-free

typedef __attribute__((ext_vector_type(16))) __bf16 v16bf;
typedef __attribute__((ext_vector_type(8)))  float  v8f;

union Frag {
  v16bf v;
  __bf16 h[16];
  uint4  q[2];
};

__device__ __forceinline__ v8f wmma_bf16(v16bf a, v16bf b, v8f c) {
  // D = A(16x32 bf16) * B(32x16 bf16) + C(16x16 f32)
  return __builtin_amdgcn_wmma_f32_16x16x32_bf16(
      /*neg_a=*/false, a, /*neg_b=*/false, b,
      /*c_mod=*/(short)0, c, /*reuse_a=*/false, /*reuse_b=*/false);
}

// --- async global->LDS (CDNA5): guarded, with sync fallback -----------------
#if defined(__has_builtin)
#if __has_builtin(__builtin_amdgcn_global_load_async_to_lds_b128)
#define HAVE_ASYNC_LDS 1
#endif
#endif

#if defined(HAVE_ASYNC_LDS)
// Builtin prototype (from hipcc diagnostic): (v4i AS(1)*, v4i AS(3)*, Ii, Ii)
typedef int v4i __attribute__((vector_size(16)));
typedef __attribute__((address_space(1))) v4i gv4i;
typedef __attribute__((address_space(3))) v4i lv4i;
__device__ __forceinline__ void async_cp16(const void* g, void* l) {
  // generic->AS casts via uintptr_t: AS(1) is numerically identical, AS(3)
  // is the low 32 bits of the generic LDS address (ISA 10.2 aperture rule).
  __builtin_amdgcn_global_load_async_to_lds_b128(
      (gv4i*)(uintptr_t)g, (lv4i*)(uintptr_t)l, 0, 0);
}
#if __has_builtin(__builtin_amdgcn_s_wait_asynccnt)
#define WAIT_ASYNC(n) __builtin_amdgcn_s_wait_asynccnt(n)
#else
#define WAIT_ASYNC(n) asm volatile("s_wait_asynccnt %0" ::"i"(n) : "memory")
#endif
#endif

// ---------------------------------------------------------------------------
// Kernel 1: per-code prep. Split embed into bf16 hi/lo, compute ||e||^2,
// and initialize the EMA output regions with the DECAY-scaled old state so
// the main kernel can accumulate (1-DECAY)-scaled stats with atomics.
// grid = KCODES blocks, block = DIMD threads.
// ---------------------------------------------------------------------------
__global__ void vq_prep(const float* __restrict__ embed,
                        const float* __restrict__ cluster_size,
                        const float* __restrict__ embed_avg,
                        __bf16* __restrict__ ehi,
                        __bf16* __restrict__ elo,
                        float* __restrict__ e2,
                        float* __restrict__ ncs,
                        float* __restrict__ nea) {
  const int k = blockIdx.x;
  const int d = threadIdx.x;
  const int idx = k * DIMD + d;

  const float v = embed[idx];
  const __bf16 hi = (__bf16)v;
  const __bf16 lo = (__bf16)(v - (float)hi);
  ehi[idx] = hi;
  elo[idx] = lo;

  nea[idx] = DECAY * embed_avg[idx];
  if (d == 0) ncs[k] = DECAY * cluster_size[k];

  __shared__ float red[DIMD];
  red[d] = v * v;
  __syncthreads();
  for (int s = DIMD / 2; s > 0; s >>= 1) {
    if (d < s) red[d] += red[d + s];
    __syncthreads();
  }
  if (d == 0) e2[k] = red[0];
}

// ---------------------------------------------------------------------------
// Kernel 2: fused distance GEMM + argmax + gather + EMA scatter.
// block = 256 threads = 8 waves; each wave owns 16 rows of x (A resident in
// VGPRs), B tiles double-buffered through LDS. grid = NROWS/128 = 375 blocks.
// ---------------------------------------------------------------------------
__global__ __launch_bounds__(256) void vq_main(
    const float* __restrict__ x,
    const float* __restrict__ embed,
    const __bf16* __restrict__ ehi,
    const __bf16* __restrict__ elo,
    const float* __restrict__ e2,
    float* __restrict__ quantize,
    float* __restrict__ codes,
    float* __restrict__ ncs,
    float* __restrict__ nea) {
  __shared__ __align__(16) unsigned short Bhi[2][16][BPITCH];
  __shared__ __align__(16) unsigned short Blo[2][16][BPITCH];
  __shared__ int codes_lds[8][16];

  const int wave = threadIdx.x >> 5;
  const int lane = threadIdx.x & 31;
  const int g    = lane >> 4;   // lane half (0 or 1)
  const int ln   = lane & 15;
  const int rowBase = blockIdx.x * 128 + wave * 16;

  // --- Build resident A fragments: 16 rows x 256 dims, bf16 hi + lo -------
  // ISA 16-bit A layout (16x32): lane holds row M = lane%16;
  // halves 0..7  <- K = 8*(lane/16) + 0..7
  // halves 8..15 <- K = 16 + 8*(lane/16) + 0..7
  Frag a_hi[8], a_lo[8];
  {
    const float* xr = x + (rowBase + ln) * DIMD;
#pragma unroll
    for (int c = 0; c < 8; ++c) {
      const int d0 = c * 32;
      const float4* p0 = (const float4*)(xr + d0 + 8 * g);
      const float4* p1 = (const float4*)(xr + d0 + 16 + 8 * g);
      float f[16];
      float4 t;
      t = p0[0]; f[0]  = t.x; f[1]  = t.y; f[2]  = t.z; f[3]  = t.w;
      t = p0[1]; f[4]  = t.x; f[5]  = t.y; f[6]  = t.z; f[7]  = t.w;
      t = p1[0]; f[8]  = t.x; f[9]  = t.y; f[10] = t.z; f[11] = t.w;
      t = p1[1]; f[12] = t.x; f[13] = t.y; f[14] = t.z; f[15] = t.w;
#pragma unroll
      for (int i = 0; i < 16; ++i) {
        const __bf16 hi = (__bf16)f[i];
        a_hi[c].h[i] = hi;
        a_lo[c].h[i] = (__bf16)(f[i] - (float)hi);
      }
    }
  }

  // --- Cooperative B-tile staging: 16 codes x 256 dims, hi + lo -----------
  // thread t copies 16-byte chunks {q, q+16} of row (t>>4) for hi and lo:
  // 4 async instructions per wave per tile -> s_wait_asynccnt 4 ensures the
  // previous tile is complete after issuing the next one (in-order).
  auto stage = [&](int k0, int buf) {
    const int row = threadIdx.x >> 4;
    const int q   = threadIdx.x & 15;
    const __bf16* gh = ehi + (k0 + row) * DIMD;
    const __bf16* gl = elo + (k0 + row) * DIMD;
#if defined(HAVE_ASYNC_LDS)
    async_cp16(gh + q * 8,        &Bhi[buf][row][q * 8]);
    async_cp16(gh + (q + 16) * 8, &Bhi[buf][row][(q + 16) * 8]);
    async_cp16(gl + q * 8,        &Blo[buf][row][q * 8]);
    async_cp16(gl + (q + 16) * 8, &Blo[buf][row][(q + 16) * 8]);
#else
    *(uint4*)&Bhi[buf][row][q * 8]        = *(const uint4*)(gh + q * 8);
    *(uint4*)&Bhi[buf][row][(q + 16) * 8] = *(const uint4*)(gh + (q + 16) * 8);
    *(uint4*)&Blo[buf][row][q * 8]        = *(const uint4*)(gl + q * 8);
    *(uint4*)&Blo[buf][row][(q + 16) * 8] = *(const uint4*)(gl + (q + 16) * 8);
#endif
  };

  // --- Sweep codes: running argmax of 2*(x.e) - ||e||^2 -------------------
  float best[8];
  int   bidx[8];
#pragma unroll
  for (int r = 0; r < 8; ++r) { best[r] = -3.4e38f; bidx[r] = 0; }

  stage(0, 0);

  const int NT = KCODES / 16;  // 128 tiles
  for (int i = 0; i < NT; ++i) {
    const int buf = i & 1;
    const int k0 = i * 16;
    if (i + 1 < NT) {
      stage(k0 + 16, buf ^ 1);
#if defined(HAVE_ASYNC_LDS)
      WAIT_ASYNC(4);   // current tile's 4 async ops complete
#endif
    } else {
#if defined(HAVE_ASYNC_LDS)
      WAIT_ASYNC(0);
#endif
    }
    __syncthreads();   // publish tile across waves

    v8f acc = {0.f, 0.f, 0.f, 0.f, 0.f, 0.f, 0.f, 0.f};
#pragma unroll
    for (int c = 0; c < 8; ++c) {
      const int d0 = c * 32;
      Frag bh, bl;
      bh.q[0] = *(const uint4*)&Bhi[buf][ln][d0 + 8 * g];
      bh.q[1] = *(const uint4*)&Bhi[buf][ln][d0 + 16 + 8 * g];
      bl.q[0] = *(const uint4*)&Blo[buf][ln][d0 + 8 * g];
      bl.q[1] = *(const uint4*)&Blo[buf][ln][d0 + 16 + 8 * g];
      // fp32-faithful 3-pass bf16 product
      acc = wmma_bf16(a_hi[c].v, bh.v, acc);
      acc = wmma_bf16(a_lo[c].v, bh.v, acc);
      acc = wmma_bf16(a_hi[c].v, bl.v, acc);
    }
    const float e2l = e2[k0 + ln];
#pragma unroll
    for (int r = 0; r < 8; ++r) {
      const float s = 2.0f * acc[r] - e2l;
      if (s > best[r]) { best[r] = s; bidx[r] = k0 + ln; }  // strict: keeps lowest k
    }
    __syncthreads();   // all waves done with buf before it is overwritten
  }

  // --- Cross-lane argmax over the 16 columns of each half -----------------
  // C/D layout: VGPR r holds (M = r + 8*(lane/16), N = lane%16).
#pragma unroll
  for (int m = 1; m < 16; m <<= 1) {
#pragma unroll
    for (int r = 0; r < 8; ++r) {
      const float ov = __shfl_xor(best[r], m, 32);
      const int   oi = __shfl_xor(bidx[r], m, 32);
      if (ov > best[r] || (ov == best[r] && oi < bidx[r])) {
        best[r] = ov; bidx[r] = oi;
      }
    }
  }
  if (ln == 0) {
#pragma unroll
    for (int r = 0; r < 8; ++r) codes_lds[wave][g * 8 + r] = bidx[r];
  }
  __syncthreads();

  // --- Epilogue: codes, embedding gather, EMA scatter ---------------------
#pragma unroll 1
  for (int r2 = 0; r2 < 16; ++r2) {
    const int k = codes_lds[wave][r2];
    const int n = rowBase + r2;
    if (lane == 0) {
      codes[n] = (float)k;
      atomicAdd(&ncs[k], OMD);
    }
    const float* er = embed + k * DIMD;
    const float* xr = x + n * DIMD;
    float* qr = quantize + n * DIMD;
#pragma unroll
    for (int d = lane; d < DIMD; d += 32) {
      qr[d] = er[d];
      atomicAdd(&nea[k * DIMD + d], OMD * xr[d]);
    }
  }
}

// ---------------------------------------------------------------------------
// Kernel 3: n = sum(new_cluster_size); inv_sm[k] = 1 / smoothed[k].
// single block of 1024 threads (2 codes each).
// ---------------------------------------------------------------------------
__global__ void vq_smooth(const float* __restrict__ ncs,
                          float* __restrict__ inv_sm) {
  __shared__ float red[1024];
  __shared__ float nshare;
  const int t = threadIdx.x;
  red[t] = ncs[t] + ncs[t + 1024];
  __syncthreads();
  for (int s = 512; s > 0; s >>= 1) {
    if (t < s) red[t] += red[t + s];
    __syncthreads();
  }
  if (t == 0) nshare = red[0];
  __syncthreads();
  const float n = nshare;
  const float scale = (n + (float)KCODES * EPSV) / n;  // smoothed = (c+eps)/(n+K*eps)*n
  inv_sm[t]        = scale / (ncs[t] + EPSV);
  inv_sm[t + 1024] = scale / (ncs[t + 1024] + EPSV);
}

// ---------------------------------------------------------------------------
// Kernel 4: new_embed = expired ? x[perm[k]] : new_embed_avg / smoothed.
// Deterministic coprime-stride "permutation" (gcd(40507, 48000) = 1).
// ---------------------------------------------------------------------------
__global__ void vq_finalize(const float* __restrict__ x,
                            const float* __restrict__ nea,
                            const float* __restrict__ ncs,
                            const float* __restrict__ inv_sm,
                            float* __restrict__ new_embed) {
  const int idx = blockIdx.x * blockDim.x + threadIdx.x;
  const int k = idx >> 8;
  const int d = idx & (DIMD - 1);
  float en = nea[idx] * inv_sm[k];
  if (ncs[k] < THRESH) {
    const long p = ((long)k * 40507L + 12345L) % (long)NROWS;
    en = x[p * DIMD + d];
  }
  new_embed[idx] = en;
}

// ---------------------------------------------------------------------------
extern "C" void kernel_launch(void* const* d_in, const int* in_sizes, int n_in,
                              void* d_out, int out_size, void* d_ws, size_t ws_size,
                              hipStream_t stream) {
  (void)in_sizes; (void)n_in; (void)out_size; (void)ws_size;

  const float* x  = (const float*)d_in[0];   // [48000, 256]
  const float* em = (const float*)d_in[1];   // [2048, 256]
  const float* cs = (const float*)d_in[2];   // [2048]
  const float* ea = (const float*)d_in[3];   // [2048, 256]

  float* out      = (float*)d_out;
  float* quantize = out;                                  // 48000*256
  float* codes    = quantize + (size_t)NROWS * DIMD;      // 48000
  float* ncs      = codes + NROWS;                        // 2048
  float* nea      = ncs + KCODES;                         // 2048*256
  float* ne       = nea + (size_t)KCODES * DIMD;          // 2048*256

  char* ws = (char*)d_ws;
  __bf16* ehi = (__bf16*)ws;                                        // 1 MB
  __bf16* elo = (__bf16*)(ws + (size_t)KCODES * DIMD * 2);          // 1 MB
  float*  e2  = (float*)(ws + (size_t)KCODES * DIMD * 4);           // 8 KB
  float*  ism = (float*)(ws + (size_t)KCODES * DIMD * 4 + KCODES * 4);

  vq_prep<<<KCODES, DIMD, 0, stream>>>(em, cs, ea, ehi, elo, e2, ncs, nea);
  vq_main<<<NROWS / 128, 256, 0, stream>>>(x, em, ehi, elo, e2,
                                           quantize, codes, ncs, nea);
  vq_smooth<<<1, 1024, 0, stream>>>(ncs, ism);
  vq_finalize<<<(KCODES * DIMD) / 256, 256, 0, stream>>>(x, nea, ncs, ism, ne);
}
